// NormalBlock_16123307229657
// MI455X (gfx1250) — 8-bit matrix operand order repaired and checked by reading, compiled, not run on hardware
//
#include <hip/hip_runtime.h>


typedef int   v8i __attribute__((ext_vector_type(8)));
typedef float v8f __attribute__((ext_vector_type(8)));

#define CCH 32
#define HH  160
#define WW  160
#define HW  (HH * WW)
#define KP3 320
#define LCOL 162

__device__ __forceinline__ int sgn8(float v) {
    return (v > 0.f) ? 1 : ((v < 0.f) ? -1 : 0);
}
__device__ __forceinline__ unsigned pack4(int a, int b, int c, int d) {
    return (unsigned)(a & 0xFF) | ((unsigned)(b & 0xFF) << 8) |
           ((unsigned)(c & 0xFF) << 16) | ((unsigned)(d & 0xFF) << 24);
}

__global__ __launch_bounds__(320)
void fracbnn_block_kernel(const float* __restrict__ x,  const float* __restrict__ w3,
                          const float* __restrict__ g3, const float* __restrict__ b3,
                          const float* __restrict__ m3, const float* __restrict__ v3,
                          const float* __restrict__ w1, const float* __restrict__ g1,
                          const float* __restrict__ b1, const float* __restrict__ m1,
                          const float* __restrict__ v1, float* __restrict__ out)
{
    __shared__ unsigned sx[3 * LCOL * CCH / 4];
    __shared__ unsigned sw3s[CCH * KP3 / 4];
    __shared__ unsigned sw1s[CCH * 64 / 4];
    __shared__ unsigned sy[10 * 16 * CCH / 4];

    const int tid = threadIdx.x;
    const int n   = blockIdx.x / HH;
    const int h   = blockIdx.x % HH;

    for (int i = tid; i < 3 * LCOL * (CCH / 4); i += 320) {
        int cig = i & 7;
        int col = (i >> 3) % LCOL;
        int row = (i >> 3) / LCOL;
        int gh = h + row - 1, gw = col - 1;
        unsigned wv = 0;
        if (gh >= 0 && gh < HH && gw >= 0 && gw < WW) {
            const float* p = x + ((size_t)n * CCH + cig * 4) * HW + (size_t)gh * WW + gw;
            wv = pack4(sgn8(p[0]), sgn8(p[HW]), sgn8(p[2 * HW]), sgn8(p[3 * HW]));
        }
        sx[i] = wv;
    }
    for (int i = tid; i < CCH * (KP3 / 4); i += 320) {
        int co = i / (KP3 / 4);
        int k4 = (i % (KP3 / 4)) * 4;
        int bt[4];
        #pragma unroll
        for (int b = 0; b < 4; ++b) {
            int K = k4 + b, p = K >> 5, ci = K & 31;
            bt[b] = (p < 9) ? sgn8(w3[co * 288 + ci * 9 + p]) : 0;
        }
        sw3s[i] = pack4(bt[0], bt[1], bt[2], bt[3]);
    }
    for (int i = tid; i < CCH * 16; i += 320) {
        int co = i / 16;
        int k4 = (i % 16) * 4;
        unsigned wv = 0;
        if (k4 < 32)
            wv = pack4(sgn8(w1[co * 32 + k4 + 0]), sgn8(w1[co * 32 + k4 + 1]),
                       sgn8(w1[co * 32 + k4 + 2]), sgn8(w1[co * 32 + k4 + 3]));
        sw1s[i] = wv;
    }
    __syncthreads();

    const int wave = tid >> 5;
    const int lane = tid & 31;
    const int ml   = lane & 15;
    const int half = lane >> 4;
    const int w0   = wave * 16;

    v8i acc0 = {}; v8i acc1 = {};
    #pragma unroll
    for (int c = 0; c < 5; ++c) {
        v8i A;
        #pragma unroll
        for (int v = 0; v < 8; ++v) {
            const int base = ((v & 2) >> 1) * 16 + (v >> 2) * 32 + (v & 1) * 4;
            const int p    = (c * 64 + base) >> 5;
            unsigned wv = 0;
            if (p < 9) {
                const int kh = p / 3, kw = p % 3;
                int ci0 = (base & 31) + half * 8;
                wv = sx[((kh * LCOL + (w0 + ml + kw)) * CCH + ci0) >> 2];
            }
            A[v] = (int)wv;
        }
        #pragma unroll
        for (int t = 0; t < 2; ++t) {
            v8i B;
            #pragma unroll
            for (int v = 0; v < 8; ++v) {
                int K = c * 64 + (v >> 2) * 32 + ((v & 2) >> 1) * 16 + half * 8 + (v & 1) * 4;
                B[v] = (int)sw3s[(((t * 16 + ml) * KP3) + K) >> 2];
            }
            if (t == 0)
                acc0 = __builtin_amdgcn_wmma_i32_16x16x64_iu8(true, A, true, B, acc0, false, false);
            else
                acc1 = __builtin_amdgcn_wmma_i32_16x16x64_iu8(true, A, true, B, acc1, false, false);
        }
    }

    const float* xrow = x + (size_t)n * CCH * HW + (size_t)h * WW;
    char* syb = (char*)sy;
    v8f y0, y1;
    #pragma unroll
    for (int t = 0; t < 2; ++t) {
        const int ch = t * 16 + ml;
        float sc = g3[ch] * rsqrtf(v3[ch] + 1e-5f);
        float sh = b3[ch] - m3[ch] * sc;
        #pragma unroll
        for (int r = 0; r < 8; ++r) {
            int pixl = half * 8 + r;
            float xv = xrow[(size_t)ch * HW + (w0 + pixl)];
            float yv = sc * (float)((t == 0 ? acc0 : acc1)[r]) + sh + xv;
            if (t == 0) y0[r] = yv; else y1[r] = yv;
            syb[(wave * 16 + pixl) * CCH + ch] = (char)sgn8(yv);
        }
    }

    v8i A1;
    #pragma unroll
    for (int v = 0; v < 8; ++v) {
        if (v < 4) {
            int koff = ((v & 2) >> 1) * 16 + (v & 1) * 4 + half * 8;
            A1[v] = (int)sy[(((wave * 16 + ml) * CCH) + koff) >> 2];
        } else {
            A1[v] = 0;
        }
    }
    v8i z0 = {}; v8i z1 = {};
    #pragma unroll
    for (int t = 0; t < 2; ++t) {
        v8i B;
        #pragma unroll
        for (int v = 0; v < 8; ++v) {
            int K = (v >> 2) * 32 + ((v & 2) >> 1) * 16 + half * 8 + (v & 1) * 4;
            B[v] = (int)sw1s[(((t * 16 + ml) * 64) + K) >> 2];
        }
        if (t == 0)
            z0 = __builtin_amdgcn_wmma_i32_16x16x64_iu8(true, A1, true, B, z0, false, false);
        else
            z1 = __builtin_amdgcn_wmma_i32_16x16x64_iu8(true, A1, true, B, z1, false, false);
    }

    float* orow = out + (size_t)n * CCH * HW + (size_t)h * WW;
    #pragma unroll
    for (int t = 0; t < 2; ++t) {
        const int ch = t * 16 + ml;
        float sc = g1[ch] * rsqrtf(v1[ch] + 1e-5f);
        float sh = b1[ch] - m1[ch] * sc;
        #pragma unroll
        for (int r = 0; r < 8; ++r) {
            int pixl = half * 8 + r;
            float yv = (t == 0) ? y0[r] : y1[r];
            float zv = sc * (float)((t == 0 ? z0 : z1)[r]) + sh + yv;
            orow[(size_t)ch * HW + (w0 + pixl)] = zv;
        }
    }
}

extern "C" void kernel_launch(void* const* d_in, const int* in_sizes, int n_in,
                              void* d_out, int out_size, void* d_ws, size_t ws_size,
                              hipStream_t stream) {
    (void)in_sizes; (void)n_in; (void)out_size; (void)d_ws; (void)ws_size;
    const float* x  = (const float*)d_in[0];
    const float* w3 = (const float*)d_in[1];
    const float* g3 = (const float*)d_in[2];
    const float* b3 = (const float*)d_in[3];
    const float* m3 = (const float*)d_in[4];
    const float* v3 = (const float*)d_in[5];
    const float* w1 = (const float*)d_in[6];
    const float* g1 = (const float*)d_in[7];
    const float* b1 = (const float*)d_in[8];
    const float* m1 = (const float*)d_in[9];
    const float* v1 = (const float*)d_in[10];
    float* out = (float*)d_out;

    dim3 grid(32 * HH);
    dim3 block(320);
    hipLaunchKernelGGL(fracbnn_block_kernel, grid, block, 0, stream,
                       x, w3, g3, b3, m3, v3, w1, g1, b1, m1, v1, out);
}
